// PoincareLinear_85177791414442
// MI455X (gfx1250) — matrix operand order repaired and checked by reading, not compiled, not run on hardware
//
#include <hip/hip_runtime.h>
#include <cstdint>
#include <cstddef>

typedef __attribute__((ext_vector_type(16))) __bf16 bf16x16;
typedef __attribute__((ext_vector_type(8)))  float  f32x8;
typedef __attribute__((ext_vector_type(4)))  float  f32x4;
typedef __attribute__((ext_vector_type(4)))  unsigned int u32x4;
typedef __attribute__((ext_vector_type(8)))  unsigned int u32x8;
typedef __attribute__((ext_vector_type(4)))  int i32x4;

// pointer-to-int4 in global (AS1) / LDS (AS3) address spaces, as required by
// __builtin_amdgcn_global_load_async_to_lds_b128 (per clang diagnostic).
typedef __attribute__((address_space(1))) i32x4* gv4p;
typedef __attribute__((address_space(3))) i32x4* lv4p;

#define EPSF 1e-5f

#if defined(__has_builtin)
#if __has_builtin(__builtin_amdgcn_global_load_async_to_lds_b128)
#define HAVE_ASYNC_LDS 1
#endif
#if __has_builtin(__builtin_amdgcn_s_wait_asynccnt)
#define WAIT_ASYNC() __builtin_amdgcn_s_wait_asynccnt(0)
#endif
#endif
#ifndef HAVE_ASYNC_LDS
#define HAVE_ASYNC_LDS 0
#endif
#ifndef WAIT_ASYNC
#define WAIT_ASYNC() asm volatile("s_wait_asynccnt 0x0" ::: "memory")
#endif

__device__ __forceinline__ unsigned short f2bf(float f) {
  unsigned u = __float_as_uint(f);
  u += 0x7FFFu + ((u >> 16) & 1u);      // round-to-nearest-even
  return (unsigned short)(u >> 16);
}

// ---------------------------------------------------------------------------
// K1a: per-row expmap0 prep: rcx = tanh(n)*x/n (bf16), cx2 = min(tanh(n)^2,1-eps)
//      also zero the per-row sum(y^2) accumulator.
// ---------------------------------------------------------------------------
__global__ __launch_bounds__(256) void prep_x(const float* __restrict__ x,
                                              unsigned short* __restrict__ rcx,
                                              float* __restrict__ cx2,
                                              float* __restrict__ ssum, int D) {
  __shared__ float red[8];
  __shared__ float shsc;
  const int row = blockIdx.x, t = threadIdx.x, w = t >> 5, lane = t & 31;
  const size_t base = (size_t)row * D;
  float ss = 0.f;
  for (int i = t; i < D; i += 256) { float v = x[base + i]; ss += v * v; }
  for (int m = 1; m <= 16; m <<= 1) ss += __shfl_xor(ss, m);
  if (lane == 0) red[w] = ss;
  __syncthreads();
  if (t == 0) {
    float tot = 0.f;
    for (int i = 0; i < 8; i++) tot += red[i];
    float nc = fmaxf(sqrtf(tot), EPSF);
    float sc = tanhf(nc) / nc;
    shsc = sc;
    cx2[row]  = fminf(sc * sc * tot, 1.f - EPSF);
    ssum[row] = 0.f;
  }
  __syncthreads();
  const float sc = shsc;
  for (int i = t; i < D; i += 256) rcx[base + i] = f2bf(x[base + i] * sc);
}

// ---------------------------------------------------------------------------
// K1b: per-output-column of z: 2*||z_col||, 1/||z_col||, cosh/sinh(2*bias)
// ---------------------------------------------------------------------------
__global__ __launch_bounds__(256) void prep_z(const float* __restrict__ z,
                                              const float* __restrict__ bias,
                                              float* __restrict__ zn2,
                                              float* __restrict__ invzn,
                                              float* __restrict__ chd,
                                              float* __restrict__ shd,
                                              int K, int N) {
  const int c = blockIdx.x * 256 + threadIdx.x;
  if (c >= N) return;
  float ss = 0.f;
  for (int k = 0; k < K; k++) { float v = z[(size_t)k * N + c]; ss += v * v; }
  float n = fmaxf(sqrtf(ss), EPSF);
  zn2[c] = 2.f * n;
  invzn[c] = 1.f / n;
  float d = 2.f * bias[c];
  chd[c] = coshf(d);
  shd[c] = sinhf(d);
}

// ---------------------------------------------------------------------------
// K1c: zt[n][k] = bf16( z[k][n] / ||z_col n|| )  (64x64 tiles through LDS)
// ---------------------------------------------------------------------------
__global__ __launch_bounds__(256) void zt_transpose(const float* __restrict__ z,
                                                    const float* __restrict__ invzn,
                                                    unsigned short* __restrict__ zt,
                                                    int K, int N) {
  __shared__ float tile[64][65];
  const int c0 = blockIdx.x * 64, k0 = blockIdx.y * 64, t = threadIdx.x;
  for (int i = 0; i < 16; i++) {
    int idx = t + i * 256;
    int kr = idx >> 6, cc = idx & 63;
    tile[cc][kr] = z[(size_t)(k0 + kr) * N + (c0 + cc)];
  }
  __syncthreads();
  for (int i = 0; i < 16; i++) {
    int idx = t + i * 256;
    int r = idx >> 6, cc = idx & 63;
    zt[(size_t)(c0 + r) * K + k0 + cc] = f2bf(tile[r][cc] * invzn[c0 + r]);
  }
}

// ---------------------------------------------------------------------------
// K2: WMMA bf16 GEMM (dot = rcx @ z_unit), async double-buffered LDS staging,
//     fused Poincare MLR epilogue:
//     y = sinh( 2*||z|| * asinh( (2*dot*cosh - (1+cx2)*sinh) / (1-cx2) ) )
// ---------------------------------------------------------------------------
#define BM 128
#define BN 128
#define BK 64
#define LDT 72   // ushort row stride (BK + 8 pad -> 144B, conflict-free b128)

__global__ __launch_bounds__(256, 1) void poincare_gemm(
    const unsigned short* __restrict__ A,   // [M][K] bf16 (rcx)
    const unsigned short* __restrict__ Bt,  // [N][K] bf16 (z_unit^T)
    const float* __restrict__ cx2, const float* __restrict__ zn2,
    const float* __restrict__ chd, const float* __restrict__ shd,
    float* __restrict__ out, float* __restrict__ ssum,
    int M, int N, int K) {
  __shared__ __align__(16) unsigned short As[2][BM * LDT];
  __shared__ __align__(16) unsigned short Bs[2][BN * LDT];

  const int t = threadIdx.x;
  const int w = t >> 5, lane = t & 31, l16 = lane & 15, half = lane >> 4;
  const int wr = w >> 1, wc = w & 1;          // 4x2 wave grid
  const int m0w = wr * 32, n0w = wc * 64;     // 32x64 sub-tile per wave
  const int blockM = blockIdx.y * BM, blockN = blockIdx.x * BN;
  const int srow = t >> 3, sch = (t & 7) * 8; // staging coords (stride 32 rows)

  f32x8 acc[2][4];
#pragma unroll
  for (int i = 0; i < 2; i++)
#pragma unroll
    for (int j = 0; j < 4; j++)
#pragma unroll
      for (int g = 0; g < 8; g++) acc[i][j][g] = 0.f;

#if HAVE_ASYNC_LDS
  auto issue = [&](int buf, int k0) {
#pragma unroll
    for (int i = 0; i < 4; i++) {
      int row = srow + i * 32;
      __builtin_amdgcn_global_load_async_to_lds_b128(
          (gv4p)(A + (size_t)(blockM + row) * K + k0 + sch),
          (lv4p)(&As[buf][row * LDT + sch]), 0, 0);
      __builtin_amdgcn_global_load_async_to_lds_b128(
          (gv4p)(Bt + (size_t)(blockN + row) * K + k0 + sch),
          (lv4p)(&Bs[buf][row * LDT + sch]), 0, 0);
    }
  };
#else
  auto issue = [&](int buf, int k0) {
#pragma unroll
    for (int i = 0; i < 4; i++) {
      int row = srow + i * 32;
      *(u32x4*)(&As[buf][row * LDT + sch]) =
          *(const u32x4*)(A + (size_t)(blockM + row) * K + k0 + sch);
      *(u32x4*)(&Bs[buf][row * LDT + sch]) =
          *(const u32x4*)(Bt + (size_t)(blockN + row) * K + k0 + sch);
    }
  };
#endif

  // prologue: fill buffer 0
  issue(0, 0);
  WAIT_ASYNC();
  __syncthreads();

  int cur = 0;
  for (int k0 = 0; k0 < K; k0 += BK) {
    if (k0 + BK < K) issue(cur ^ 1, k0 + BK);   // prefetch next tile (async)

#pragma unroll
    for (int kk = 0; kk < BK; kk += 32) {
      bf16x16 af[2], bfr[4];
      // A frag (16x32): lane half 0 -> K {kk..+7, kk+16..+23}; half 1 -> +8
#pragma unroll
      for (int fm = 0; fm < 2; fm++) {
        const unsigned short* pa =
            &As[cur][(m0w + fm * 16 + l16) * LDT + kk + half * 8];
        u32x4 lo = *(const u32x4*)pa;
        u32x4 hi = *(const u32x4*)(pa + 16);
        u32x8 cat;
#pragma unroll
        for (int q = 0; q < 4; q++) { cat[q] = lo[q]; cat[q + 4] = hi[q]; }
        af[fm] = __builtin_bit_cast(bf16x16, cat);
      }
      // B frag (32x16): lane = column, half 0 -> K {kk..+7, kk+16..+23}; half 1 -> +8
#pragma unroll
      for (int fn = 0; fn < 4; fn++) {
        const unsigned short* pb =
            &Bs[cur][(n0w + fn * 16 + l16) * LDT + kk + half * 8];
        u32x4 lo = *(const u32x4*)pb;
        u32x4 hi = *(const u32x4*)(pb + 16);
        u32x8 cat;
#pragma unroll
        for (int q = 0; q < 4; q++) { cat[q] = lo[q]; cat[q + 4] = hi[q]; }
        bfr[fn] = __builtin_bit_cast(bf16x16, cat);
      }
#pragma unroll
      for (int fm = 0; fm < 2; fm++)
#pragma unroll
        for (int fn = 0; fn < 4; fn++)
          acc[fm][fn] = __builtin_amdgcn_wmma_f32_16x16x32_bf16(
              false, af[fm], false, bfr[fn], (short)0, acc[fm][fn], false, false);
    }

    WAIT_ASYNC();       // our async writes to the other buffer are done
    __syncthreads();    // everyone finished reading cur + their asyncs landed
    cur ^= 1;
  }

  // ---- fused Poincare MLR epilogue (lean: no big live arrays) ----------
  float chv[4], shv[4], zzv[4];
#pragma unroll
  for (int fn = 0; fn < 4; fn++) {
    int c = blockN + n0w + fn * 16 + l16;
    chv[fn] = chd[c]; shv[fn] = shd[c]; zzv[fn] = zn2[c];
  }
#pragma unroll
  for (int fm = 0; fm < 2; fm++)
#pragma unroll
    for (int g = 0; g < 8; g++) {
      const int r = blockM + m0w + fm * 16 + half * 8 + g;
      const float cx = cx2[r];
      const float onep = 1.f + cx;
      const float invd = 1.f / (1.f - cx);
      float rowsq = 0.f;
#pragma unroll
      for (int fn = 0; fn < 4; fn++) {
        float u = (2.f * acc[fm][fn][g] * chv[fn] - onep * shv[fn]) * invd;
        float m = zzv[fn] * __logf(u + __builtin_sqrtf(u * u + 1.f)); // asinh
        float E = __expf(m);
        float y = 0.5f * (E - 1.f / E);                               // sinh
        out[(size_t)r * N + (blockN + n0w + fn * 16 + l16)] = y;
        rowsq += y * y;
      }
      // 16-lane reduction per half-wave (halves hold different rows)
      rowsq += __shfl_xor(rowsq, 1);
      rowsq += __shfl_xor(rowsq, 2);
      rowsq += __shfl_xor(rowsq, 4);
      rowsq += __shfl_xor(rowsq, 8);
      if (l16 == 0) atomicAdd(&ssum[r], rowsq);
    }
}

// ---------------------------------------------------------------------------
// K3: in-place per-row scale implementing ball-projection + logmap0
// ---------------------------------------------------------------------------
__global__ __launch_bounds__(256) void finalize(float* __restrict__ out,
                                                const float* __restrict__ ssum,
                                                int N, size_t total4) {
  size_t idx = (size_t)blockIdx.x * 256 + threadIdx.x;
  if (idx >= total4) return;
  size_t e = idx * 4;
  int row = (int)(e / (size_t)N);
  float s = ssum[row];
  float denom = 1.f + sqrtf(1.f + s);
  float nb = sqrtf(s) / denom;
  float ncl = fmaxf(nb, EPSF);
  float tcl = fminf(ncl, 1.f - EPSF);
  float f = (0.5f * __logf((1.f + tcl) / (1.f - tcl))) / (ncl * denom); // atanh
  f32x4* p = (f32x4*)(out + e);
  f32x4 v = *p;
  v *= f;
  *p = v;
}

// ---------------------------------------------------------------------------
extern "C" void kernel_launch(void* const* d_in, const int* in_sizes, int n_in,
                              void* d_out, int out_size, void* d_ws, size_t ws_size,
                              hipStream_t stream) {
  const float* x    = (const float*)d_in[0];
  const float* z    = (const float*)d_in[1];
  const float* bias = (const float*)d_in[2];
  const int DOUT = in_sizes[2];
  const int DIN  = in_sizes[1] / DOUT;
  const int B    = in_sizes[0] / DIN;
  float* out = (float*)d_out;

  char* ws = (char*)d_ws;
  unsigned short* rcx = (unsigned short*)ws;  ws += (size_t)B * DIN * 2;
  unsigned short* zt  = (unsigned short*)ws;  ws += (size_t)DIN * DOUT * 2;
  float* cx2   = (float*)ws; ws += (size_t)B * 4;
  float* ssum  = (float*)ws; ws += (size_t)B * 4;
  float* zn2   = (float*)ws; ws += (size_t)DOUT * 4;
  float* invzn = (float*)ws; ws += (size_t)DOUT * 4;
  float* chd   = (float*)ws; ws += (size_t)DOUT * 4;
  float* shd   = (float*)ws; ws += (size_t)DOUT * 4;

  prep_x<<<B, 256, 0, stream>>>(x, rcx, cx2, ssum, DIN);
  prep_z<<<(DOUT + 255) / 256, 256, 0, stream>>>(z, bias, zn2, invzn, chd, shd,
                                                 DIN, DOUT);
  zt_transpose<<<dim3(DOUT / 64, DIN / 64), 256, 0, stream>>>(z, invzn, zt,
                                                              DIN, DOUT);
  poincare_gemm<<<dim3(DOUT / BN, B / BM), 256, 0, stream>>>(
      rcx, zt, cx2, zn2, chd, shd, out, ssum, B, DOUT, DIN);
  size_t total4 = (size_t)B * DOUT / 4;
  finalize<<<(total4 + 255) / 256, 256, 0, stream>>>(out, ssum, DOUT, total4);
}
